// ami_only_net2_82832739270726
// MI455X (gfx1250) — matrix operand order repaired and checked by reading, not compiled, not run on hardware
//
#include <hip/hip_runtime.h>
#include <cstdint>

// ---------------- problem constants (match reference) ----------------
#define NN    100000   // nodes
#define EE    1600000  // edges
#define BB    256      // batch / graphs
#define FIN   52
#define DD1   128
#define DD2   64
#define LEMB  1900
#define L1    948      // after conv1(K=5)+pool2
#define L2V   473      // after conv2(K=3)+pool2
#define SEQF  9460     // 20*473
#define SEQFP 9472     // padded to mult of 32
#define ENTOT (EE + NN)  // edges + self loops

typedef __attribute__((ext_vector_type(16))) __bf16 v16bfx;
typedef __attribute__((ext_vector_type(8)))  float  v8fx;

union FragBF { uint4 q[2]; v16bfx v; };

// ---------------- small utility kernels ----------------
__global__ void k_fill(float* p, float v, int n) {
  int t = blockIdx.x * blockDim.x + threadIdx.x;
  if (t < n) p[t] = v;
}

// embedding [B,1,64,1900] -> max over dim2 -> [B,1900]
__global__ void k_embed_max(const float* __restrict__ emb, float* __restrict__ out) {
  int t = blockIdx.x * blockDim.x + threadIdx.x;
  if (t >= BB * LEMB) return;
  int l = t % LEMB, b = t / LEMB;
  const float* e = emb + (size_t)b * 64 * LEMB + l;
  float m = -__builtin_inff();
  #pragma unroll 4
  for (int h = 0; h < 64; ++h) m = fmaxf(m, e[(size_t)h * LEMB]);
  out[t] = m;
}

// conv1d(1->10,K=5) + relu + maxpool2 : emb[B,1900] -> a1[B,10,948]
__global__ void k_conv1(const float* __restrict__ emb, const float* __restrict__ w,
                        const float* __restrict__ bias, float* __restrict__ out) {
  int t = blockIdx.x * blockDim.x + threadIdx.x;
  if (t >= BB * 10 * L1) return;
  int l = t % L1, u = t / L1, co = u % 10, b = u / 10;
  const float* e = emb + (size_t)b * LEMB + 2 * l;
  const float* wk = w + co * 5;
  float v0 = bias[co], v1 = v0;
  #pragma unroll
  for (int k = 0; k < 5; ++k) { v0 += e[k] * wk[k]; v1 += e[k + 1] * wk[k]; }
  out[t] = fmaxf(fmaxf(v0, 0.f), fmaxf(v1, 0.f));
}

// conv1d(10->20,K=3) + relu + maxpool2 : a1[B,10,948] -> seq[B, 20*473]
__global__ void k_conv2(const float* __restrict__ a1, const float* __restrict__ w,
                        const float* __restrict__ bias, float* __restrict__ seq) {
  int t = blockIdx.x * blockDim.x + threadIdx.x;
  if (t >= BB * 20 * L2V) return;
  int l = t % L2V, u = t / L2V, co = u % 20, b = u / 20;
  float v0 = bias[co], v1 = v0;
  const float* ab = a1 + (size_t)b * 10 * L1 + 2 * l;
  #pragma unroll
  for (int ci = 0; ci < 10; ++ci) {
    const float* r = ab + (size_t)ci * L1;
    const float* wk = w + (co * 10 + ci) * 3;
    v0 += r[0] * wk[0] + r[1] * wk[1] + r[2] * wk[2];
    v1 += r[1] * wk[0] + r[2] * wk[1] + r[3] * wk[2];
  }
  seq[(size_t)b * SEQF + co * L2V + l] = fmaxf(fmaxf(v0, 0.f), fmaxf(v1, 0.f));
}

// f32 [M,K] -> bf16 [M,Kp]
__global__ void k_cvt_rows(const float* __restrict__ in, __bf16* __restrict__ out,
                           int M, int K, int Kp) {
  size_t t = (size_t)blockIdx.x * blockDim.x + threadIdx.x;
  if (t >= (size_t)M * Kp) return;
  int kp = (int)(t % Kp); size_t m = t / Kp;
  out[t] = (kp < K) ? (__bf16)in[m * K + kp] : (__bf16)0.f;
}

// f32 [K,N] -> bf16 transposed [N,Kp]
__global__ void k_cvt_trans(const float* __restrict__ in, __bf16* __restrict__ out,
                            int K, int N, int Kp) {
  size_t t = (size_t)blockIdx.x * blockDim.x + threadIdx.x;
  if (t >= (size_t)N * Kp) return;
  int kp = (int)(t % Kp); size_t n = t / Kp;
  out[t] = (kp < K) ? (__bf16)in[(size_t)kp * N + n] : (__bf16)0.f;
}

// ---------------- WMMA fragment helpers (contiguous k order, same for A and B) ----------------
// A: lane&15 = M-row; elems 0..15 -> K=k0+16h+e (contiguous 32B)
// B (stored transposed): lane&15 = N-col; elems 0..15 -> K=k0+16h+e (contiguous 32B)
__device__ inline void load_fragA(FragBF& fa, const __bf16* Arow, int k0, int half) {
  fa.q[0] = *(const uint4*)(Arow + k0 + half * 16);
  fa.q[1] = *(const uint4*)(Arow + k0 + half * 16 + 8);
}
__device__ inline void load_fragB(FragBF& fb, const __bf16* Brow, int k0, int half) {
  const uint4* bp = (const uint4*)(Brow + k0 + half * 16);
  fb.q[0] = bp[0]; fb.q[1] = bp[1];
}

// ---------------- direct-register bf16 WMMA GEMM (node-scale GEMMs) ----------
// One wave per 16x16 tile, register double-buffered K pipeline.
__global__ void __launch_bounds__(32)
k_gemm_bf16_wmma(const __bf16* __restrict__ A, const __bf16* __restrict__ Bt,
                 const float* __restrict__ bias,
                 float* __restrict__ Cf, __bf16* __restrict__ Cb,
                 int M, int N, int Kp, int reluF, int reluB) {
  const int lane = threadIdx.x;
  const int half = lane >> 4;
  const int l15  = lane & 15;
  const int tm = blockIdx.y, tn = blockIdx.x;
  const __bf16* Arow = A  + (size_t)(tm * 16 + l15) * Kp;
  const __bf16* Brow = Bt + (size_t)(tn * 16 + l15) * Kp;
  v8fx acc = {};
  FragBF fa, fb, na, nb;
  load_fragA(fa, Arow, 0, half);
  load_fragB(fb, Brow, 0, half);
  for (int k0 = 0; k0 < Kp; k0 += 32) {
    const int kn = k0 + 32;
    if (kn < Kp) {                       // issue next step's loads before this WMMA
      load_fragA(na, Arow, kn, half);
      load_fragB(nb, Brow, kn, half);
      if (kn + 96 < Kp) {                // global_prefetch_b8, 4 steps ahead
        __builtin_prefetch(Arow + kn + 96, 0, 1);
        __builtin_prefetch(Brow + kn + 96, 0, 1);
      }
    }
    acc = __builtin_amdgcn_wmma_f32_16x16x32_bf16(
        false, fa.v, false, fb.v, (short)0, acc, false, false);
    fa = na; fb = nb;
  }
  const int col = tn * 16 + l15;
  const float bv = bias ? bias[col] : 0.f;
  #pragma unroll
  for (int r = 0; r < 8; ++r) {
    const int row = tm * 16 + half * 8 + r;
    const float v = acc[r] + bv;
    if (Cf) Cf[(size_t)row * N + col] = reluF ? fmaxf(v, 0.f) : v;
    if (Cb) Cb[(size_t)row * N + col] = (__bf16)(reluB ? fmaxf(v, 0.f) : v);
  }
}

// ---------------- async-LDS bf16 WMMA GEMM (FC GEMMs, M multiple of 64) ------
// 4 waves/block; block computes 64x16 output. The shared 16-col B tile for each
// K-step is staged into LDS with the CDNA5 async DMA path
// (global_load_async_to_lds_b128, ASYNCcnt), double-buffered so the copy for
// step k+1 overlaps the WMMA of step k. A fragments stream from L2 directly.
__device__ inline void async_stage_b(const __bf16* __restrict__ Bt, size_t rowbase,
                                     int Kp, int k0, __bf16* tile, int lane) {
  #pragma unroll
  for (int h = 0; h < 2; ++h) {
    const int j = lane + h * 32;              // 64 chunks of 16B = 1KB tile
    const int c = j >> 2, p = j & 3;          // col c (0..15), 8-elem part p
    const __bf16* g = Bt + rowbase + (size_t)c * Kp + k0 + p * 8;
    const unsigned loff = (unsigned)(size_t)&tile[c * 32 + p * 8];  // LDS byte off
    asm volatile("global_load_async_to_lds_b128 %0, %1, off"
                 :: "v"(loff), "v"(g) : "memory");
  }
}

__global__ void __launch_bounds__(128)
k_gemm_bf16_wmma_async(const __bf16* __restrict__ A, const __bf16* __restrict__ Bt,
                       const float* __restrict__ bias,
                       float* __restrict__ Cf, __bf16* __restrict__ Cb,
                       int M, int N, int Kp, int reluF, int reluB) {
  __shared__ __align__(16) __bf16 btile[2][16 * 32];   // 2KB double buffer
  const int lane = threadIdx.x & 31;
  const int wid  = threadIdx.x >> 5;
  const int half = lane >> 4;
  const int l15  = lane & 15;
  const int tm = blockIdx.y * 4 + wid, tn = blockIdx.x;
  const __bf16* Arow = A + (size_t)(tm * 16 + l15) * Kp;
  const size_t brow = (size_t)(tn * 16) * Kp;
  v8fx acc = {};

  if (wid == 0) async_stage_b(Bt, brow, Kp, 0, &btile[0][0], lane);

  int buf = 0;
  for (int k0 = 0; k0 < Kp; k0 += 32, buf ^= 1) {
    FragBF fa;
    load_fragA(fa, Arow, k0, half);            // in flight across the barrier
    if (wid == 0) {
      if (k0 + 32 < Kp) {                      // stage next tile, then wait only
        async_stage_b(Bt, brow, Kp, k0 + 32, &btile[buf ^ 1][0], lane);
        asm volatile("s_wait_asynccnt 0x2" ::: "memory");   // current tile done
      } else {
        asm volatile("s_wait_asynccnt 0x0" ::: "memory");
      }
    }
    __syncthreads();                           // tile visible to all 4 waves
    FragBF fb;
    fb.q[0] = *(const uint4*)&btile[buf][l15 * 32 + half * 16];
    fb.q[1] = *(const uint4*)&btile[buf][l15 * 32 + half * 16 + 8];
    acc = __builtin_amdgcn_wmma_f32_16x16x32_bf16(
        false, fa.v, false, fb.v, (short)0, acc, false, false);
    __syncthreads();                           // reads done before buf re-staged
  }

  const int col = tn * 16 + l15;
  const float bv = bias ? bias[col] : 0.f;
  #pragma unroll
  for (int r = 0; r < 8; ++r) {
    const int row = tm * 16 + half * 8 + r;
    const float v = acc[r] + bv;
    if (Cf) Cf[(size_t)row * N + col] = reluF ? fmaxf(v, 0.f) : v;
    if (Cb) Cb[(size_t)row * N + col] = (__bf16)(reluB ? fmaxf(v, 0.f) : v);
  }
}

// ---------------- graph kernels ----------------
__device__ inline void edge_sd(const int* __restrict__ ei, int idx, int& s, int& d) {
  if (idx < EE) { s = ei[idx]; d = ei[EE + idx]; }
  else          { s = d = idx - EE; }               // self loops
}

__device__ inline void atomicMaxF(float* addr, float v) {
  if (v >= 0.f) atomicMax((int*)addr, __float_as_int(v));
  else          atomicMin((unsigned int*)addr, __float_as_uint(v));
}

__global__ void k_gat_scores(const float* __restrict__ h, const float* __restrict__ as,
                             const float* __restrict__ ad, float* __restrict__ osrc,
                             float* __restrict__ odst) {
  int n = blockIdx.x * blockDim.x + threadIdx.x;
  if (n >= NN) return;
  const float* hn = h + (size_t)n * DD2;
  float s = 0.f, d = 0.f;
  #pragma unroll 4
  for (int f = 0; f < DD2; ++f) { s += hn[f] * as[f]; d += hn[f] * ad[f]; }
  osrc[n] = s; odst[n] = d;
}

__global__ void k_edge_max(const int* __restrict__ ei, const float* __restrict__ asrc,
                           const float* __restrict__ adst, float* __restrict__ ebuf,
                           float* __restrict__ m) {
  int t = blockIdx.x * blockDim.x + threadIdx.x;
  if (t >= ENTOT) return;
  int s, d; edge_sd(ei, t, s, d);
  float e = asrc[s] + adst[d];
  e = (e > 0.f) ? e : 0.2f * e;                      // leaky_relu 0.2
  ebuf[t] = e;
  atomicMaxF(&m[d], e);
}

__global__ void k_edge_expsum(const int* __restrict__ ei, float* __restrict__ ebuf,
                              const float* __restrict__ m, float* __restrict__ den) {
  int t = blockIdx.x * blockDim.x + threadIdx.x;
  if (t >= ENTOT) return;
  int s, d; edge_sd(ei, t, s, d);
  float ee = __expf(ebuf[t] - m[d]);
  ebuf[t] = ee;
  atomicAdd(&den[d], ee);
}

__global__ void k_gat_aggr(const int* __restrict__ ei, const float* __restrict__ ebuf,
                           const float* __restrict__ den, const float* __restrict__ h,
                           float* __restrict__ xg) {
  int t = blockIdx.x * blockDim.x + threadIdx.x;
  if (t >= ENTOT * 16) return;
  int f4 = t & 15, idx = t >> 4;
  int s, d; edge_sd(ei, idx, s, d);
  float alpha = ebuf[idx] / den[d];
  const float* hs = h  + (size_t)s * DD2 + f4 * 4;
  float*       xo = xg + (size_t)d * DD2 + f4 * 4;
  #pragma unroll
  for (int j = 0; j < 4; ++j) atomicAdd(xo + j, alpha * hs[j]);
}

__global__ void k_gat_fin(const float* __restrict__ xg, const float* __restrict__ gb,
                          __bf16* __restrict__ xb) {
  size_t t = (size_t)blockIdx.x * blockDim.x + threadIdx.x;
  if (t >= (size_t)NN * DD2) return;
  int f = (int)(t & (DD2 - 1));
  xb[t] = (__bf16)fmaxf(xg[t] + gb[f], 0.f);
}

__global__ void k_deg(const int* __restrict__ ei, float* __restrict__ deg) {
  int t = blockIdx.x * blockDim.x + threadIdx.x;
  if (t >= ENTOT) return;
  int s, d; edge_sd(ei, t, s, d);
  atomicAdd(&deg[d], 1.0f);
}

__global__ void k_dinv(const float* __restrict__ deg, float* __restrict__ dinv) {
  int n = blockIdx.x * blockDim.x + threadIdx.x;
  if (n >= NN) return;
  dinv[n] = rsqrtf(fmaxf(deg[n], 1.0f));
}

__global__ void k_gcn_aggr(const int* __restrict__ ei, const float* __restrict__ dinv,
                           const float* __restrict__ hw, float* __restrict__ xg) {
  int t = blockIdx.x * blockDim.x + threadIdx.x;
  if (t >= ENTOT * 32) return;
  int f4 = t & 31, idx = t >> 5;
  int s, d; edge_sd(ei, idx, s, d);
  float nrm = dinv[s] * dinv[d];
  const float* hs = hw + (size_t)s * 128 + f4 * 4;
  float*       xo = xg + (size_t)d * 128 + f4 * 4;
  #pragma unroll
  for (int j = 0; j < 4; ++j) atomicAdd(xo + j, nrm * hs[j]);
}

__global__ void k_gcn_fin_pool(const float* __restrict__ xg, const float* __restrict__ gb,
                               const int* __restrict__ batch1, float* __restrict__ pooled) {
  size_t t = (size_t)blockIdx.x * blockDim.x + threadIdx.x;
  if (t >= (size_t)NN * 128) return;
  int f = (int)(t & 127); int n = (int)(t >> 7);
  float v = fmaxf(xg[t] + gb[f], 0.f);
  atomicAdd(&pooled[(size_t)batch1[n] * 128 + f], v);
}

__global__ void k_cnt(const int* __restrict__ batch1, float* __restrict__ cnt) {
  int n = blockIdx.x * blockDim.x + threadIdx.x;
  if (n >= NN) return;
  atomicAdd(&cnt[batch1[n]], 1.0f);
}

__global__ void k_concat(const float* __restrict__ aout, const float* __restrict__ pooled,
                         const float* __restrict__ cnt, __bf16* __restrict__ xcb) {
  int t = blockIdx.x * blockDim.x + threadIdx.x;
  if (t >= BB * 256) return;
  int f = t & 255, b = t >> 8;
  float v = (f < 128) ? aout[(size_t)b * 128 + f]
                      : pooled[(size_t)b * 128 + (f - 128)] / fmaxf(cnt[b], 1.0f);
  xcb[t] = (__bf16)v;
}

__global__ void k_outhead(const float* __restrict__ y2, const float* __restrict__ ow,
                          const float* __restrict__ ob, float* __restrict__ out) {
  int b = blockIdx.x * blockDim.x + threadIdx.x;
  if (b >= BB) return;
  const float* y = y2 + (size_t)b * 128;
  float s = ob[0];
  #pragma unroll 4
  for (int f = 0; f < 128; ++f) s += y[f] * ow[f];
  out[b] = s;
}

// ---------------- host launcher ----------------
static inline dim3 gblk(size_t n) { return dim3((unsigned)((n + 255) / 256)); }

extern "C" void kernel_launch(void* const* d_in, const int* in_sizes, int n_in,
                              void* d_out, int out_size, void* d_ws, size_t ws_size,
                              hipStream_t stream) {
  const float* x1      = (const float*)d_in[0];
  const float* embed   = (const float*)d_in[1];
  const float* w1      = (const float*)d_in[2];
  const float* b1      = (const float*)d_in[3];
  const float* w2      = (const float*)d_in[4];
  const float* b2      = (const float*)d_in[5];
  const float* gat_w   = (const float*)d_in[6];
  const float* gat_as  = (const float*)d_in[7];
  const float* gat_ad  = (const float*)d_in[8];
  const float* gat_b   = (const float*)d_in[9];
  const float* gcn_w   = (const float*)d_in[10];
  const float* gcn_b   = (const float*)d_in[11];
  const float* c3_w    = (const float*)d_in[12];
  const float* c3_b    = (const float*)d_in[13];
  const float* c4_w    = (const float*)d_in[14];
  const float* c4_b    = (const float*)d_in[15];
  const float* fc00_w  = (const float*)d_in[16];
  const float* fc00_b  = (const float*)d_in[17];
  const float* fc01_w  = (const float*)d_in[18];
  const float* fc01_b  = (const float*)d_in[19];
  const float* fc1_w   = (const float*)d_in[20];
  const float* fc1_b   = (const float*)d_in[21];
  const float* fc2_w   = (const float*)d_in[22];
  const float* fc2_b   = (const float*)d_in[23];
  const float* out_w   = (const float*)d_in[24];
  const float* out_b   = (const float*)d_in[25];
  const int*   eidx    = (const int*)d_in[26];
  const int*   batch1  = (const int*)d_in[27];
  float* outf = (float*)d_out;                 // [0,256) = out ; [256,...) = h2[256,512]

  char* wsp = (char*)d_ws;
  auto alloc = [&](size_t bytes) -> void* {
    void* p = (void*)wsp; wsp += (bytes + 255) & ~(size_t)255; return p;
  };
  float*  emb_max  = (float*)alloc((size_t)BB * LEMB * 4);
  float*  a1       = (float*)alloc((size_t)BB * 10 * L1 * 4);
  float*  seq      = (float*)alloc((size_t)BB * SEQF * 4);
  __bf16* seq_b    = (__bf16*)alloc((size_t)BB * SEQFP * 2);
  __bf16* fc00_wt  = (__bf16*)alloc((size_t)1024 * SEQFP * 2);
  __bf16* fc00_ob  = (__bf16*)alloc((size_t)BB * 1024 * 2);
  __bf16* fc01_wt  = (__bf16*)alloc((size_t)128 * 1024 * 2);
  float*  a_out    = (float*)alloc((size_t)BB * 128 * 4);
  __bf16* x1_b     = (__bf16*)alloc((size_t)NN * 64 * 2);
  __bf16* w1t      = (__bf16*)alloc((size_t)128 * 64 * 2);
  __bf16* xh_b     = (__bf16*)alloc((size_t)NN * 128 * 2);
  __bf16* w2t      = (__bf16*)alloc((size_t)64 * 128 * 2);
  __bf16* x64_b    = (__bf16*)alloc((size_t)NN * 64 * 2);
  __bf16* gat_wt   = (__bf16*)alloc((size_t)64 * 64 * 2);
  float*  hmat     = (float*)alloc((size_t)NN * 64 * 4);
  float*  asrc     = (float*)alloc((size_t)NN * 4);
  float*  adst     = (float*)alloc((size_t)NN * 4);
  float*  mmax     = (float*)alloc((size_t)NN * 4);
  float*  ebuf     = (float*)alloc((size_t)ENTOT * 4);
  float*  den      = (float*)alloc((size_t)NN * 4);
  float*  xgat     = (float*)alloc((size_t)NN * 64 * 4);
  __bf16* xgat_b   = (__bf16*)alloc((size_t)NN * 64 * 2);
  __bf16* gcn_wt   = (__bf16*)alloc((size_t)128 * 64 * 2);
  float*  hw       = (float*)alloc((size_t)NN * 128 * 4);
  float*  deg      = (float*)alloc((size_t)NN * 4);
  float*  dinv     = (float*)alloc((size_t)NN * 4);
  float*  xgcn     = (float*)alloc((size_t)NN * 128 * 4);
  float*  pooled   = (float*)alloc((size_t)BB * 128 * 4);
  float*  cnt      = (float*)alloc((size_t)BB * 4);
  __bf16* xc_b     = (__bf16*)alloc((size_t)BB * 256 * 2);
  __bf16* fc1_wt   = (__bf16*)alloc((size_t)512 * 256 * 2);
  __bf16* y_b      = (__bf16*)alloc((size_t)BB * 512 * 2);
  __bf16* fc2_wt   = (__bf16*)alloc((size_t)128 * 512 * 2);
  float*  y2       = (float*)alloc((size_t)BB * 128 * 4);
  (void)ws_size; (void)n_in; (void)in_sizes; (void)out_size;

  dim3 B256(256);

  // ---- sequence branch ----
  k_embed_max<<<gblk((size_t)BB * LEMB), B256, 0, stream>>>(embed, emb_max);
  k_conv1<<<gblk((size_t)BB * 10 * L1), B256, 0, stream>>>(emb_max, c3_w, c3_b, a1);
  k_conv2<<<gblk((size_t)BB * 20 * L2V), B256, 0, stream>>>(a1, c4_w, c4_b, seq);
  k_cvt_rows<<<gblk((size_t)BB * SEQFP), B256, 0, stream>>>(seq, seq_b, BB, SEQF, SEQFP);
  k_cvt_trans<<<gblk((size_t)1024 * SEQFP), B256, 0, stream>>>(fc00_w, fc00_wt, SEQF, 1024, SEQFP);
  // fc00: [256,9472]x[9472,1024]+b  (async-LDS GEMM, K=9472)
  k_gemm_bf16_wmma_async<<<dim3(1024 / 16, BB / 64), dim3(128), 0, stream>>>(
      seq_b, fc00_wt, fc00_b, nullptr, fc00_ob, BB, 1024, SEQFP, 0, 0);
  k_cvt_trans<<<gblk((size_t)128 * 1024), B256, 0, stream>>>(fc01_w, fc01_wt, 1024, 128, 1024);
  k_gemm_bf16_wmma_async<<<dim3(128 / 16, BB / 64), dim3(128), 0, stream>>>(
      fc00_ob, fc01_wt, fc01_b, a_out, nullptr, BB, 128, 1024, 0, 0);

  // ---- node MLP (direct-register GEMMs, M = 100000) ----
  k_cvt_rows<<<gblk((size_t)NN * 64), B256, 0, stream>>>(x1, x1_b, NN, FIN, 64);
  k_cvt_trans<<<gblk((size_t)128 * 64), B256, 0, stream>>>(w1, w1t, FIN, DD1, 64);
  k_gemm_bf16_wmma<<<dim3(DD1 / 16, NN / 16), dim3(32), 0, stream>>>(
      x1_b, w1t, b1, nullptr, xh_b, NN, DD1, 64, 0, 1);
  k_cvt_trans<<<gblk((size_t)64 * 128), B256, 0, stream>>>(w2, w2t, DD1, DD2, 128);
  k_gemm_bf16_wmma<<<dim3(DD2 / 16, NN / 16), dim3(32), 0, stream>>>(
      xh_b, w2t, b2, nullptr, x64_b, NN, DD2, 128, 0, 1);

  // ---- GAT ----
  k_cvt_trans<<<gblk((size_t)64 * 64), B256, 0, stream>>>(gat_w, gat_wt, DD2, DD2, 64);
  k_gemm_bf16_wmma<<<dim3(DD2 / 16, NN / 16), dim3(32), 0, stream>>>(
      x64_b, gat_wt, nullptr, hmat, nullptr, NN, DD2, 64, 0, 0);
  k_gat_scores<<<gblk(NN), B256, 0, stream>>>(hmat, gat_as, gat_ad, asrc, adst);
  k_fill<<<gblk(NN), B256, 0, stream>>>(mmax, -__builtin_inff(), NN);
  k_fill<<<gblk(NN), B256, 0, stream>>>(den, 0.f, NN);
  k_fill<<<gblk((size_t)NN * 64), B256, 0, stream>>>(xgat, 0.f, NN * 64);
  k_edge_max<<<gblk(ENTOT), B256, 0, stream>>>(eidx, asrc, adst, ebuf, mmax);
  k_edge_expsum<<<gblk(ENTOT), B256, 0, stream>>>(eidx, ebuf, mmax, den);
  k_gat_aggr<<<gblk((size_t)ENTOT * 16), B256, 0, stream>>>(eidx, ebuf, den, hmat, xgat);
  k_gat_fin<<<gblk((size_t)NN * 64), B256, 0, stream>>>(xgat, gat_b, xgat_b);

  // ---- GCN ----
  k_cvt_trans<<<gblk((size_t)128 * 64), B256, 0, stream>>>(gcn_w, gcn_wt, DD2, 128, 64);
  k_gemm_bf16_wmma<<<dim3(128 / 16, NN / 16), dim3(32), 0, stream>>>(
      xgat_b, gcn_wt, nullptr, hw, nullptr, NN, 128, 64, 0, 0);
  k_fill<<<gblk(NN), B256, 0, stream>>>(deg, 0.f, NN);
  k_deg<<<gblk(ENTOT), B256, 0, stream>>>(eidx, deg);
  k_dinv<<<gblk(NN), B256, 0, stream>>>(deg, dinv);
  k_fill<<<gblk((size_t)NN * 128), B256, 0, stream>>>(xgcn, 0.f, NN * 128);
  k_gcn_aggr<<<gblk((size_t)ENTOT * 32), B256, 0, stream>>>(eidx, dinv, hw, xgcn);

  // ---- global mean pool ----
  k_fill<<<gblk((size_t)BB * 128), B256, 0, stream>>>(pooled, 0.f, BB * 128);
  k_fill<<<gblk(BB), B256, 0, stream>>>(cnt, 0.f, BB);
  k_gcn_fin_pool<<<gblk((size_t)NN * 128), B256, 0, stream>>>(xgcn, gcn_b, batch1, pooled);
  k_cnt<<<gblk(NN), B256, 0, stream>>>(batch1, cnt);

  // ---- head ----
  k_concat<<<gblk((size_t)BB * 256), B256, 0, stream>>>(a_out, pooled, cnt, xc_b);
  k_cvt_trans<<<gblk((size_t)512 * 256), B256, 0, stream>>>(fc1_w, fc1_wt, 256, 512, 256);
  k_gemm_bf16_wmma_async<<<dim3(512 / 16, BB / 64), dim3(128), 0, stream>>>(
      xc_b, fc1_wt, fc1_b, outf + BB, y_b, BB, 512, 256, 0, 1);
  k_cvt_trans<<<gblk((size_t)128 * 512), B256, 0, stream>>>(fc2_w, fc2_wt, 512, 128, 512);
  k_gemm_bf16_wmma_async<<<dim3(128 / 16, BB / 64), dim3(128), 0, stream>>>(
      y_b, fc2_wt, fc2_b, y2, nullptr, BB, 128, 512, 1, 0);
  k_outhead<<<gblk(BB), B256, 0, stream>>>(y2, out_w, out_b, outf);
}
